// SparseFFN_69750268887538
// MI455X (gfx1250) — matrix operand order repaired and checked by reading, not compiled, not run on hardware
//
#include <hip/hip_runtime.h>
#include <cstdint>

#define T_TOK 8192
#define DMODEL 1024
#define NEXP 7
#define KTOP 2
#define CAP 2340   // int(2.0 * 8192 / 7)

typedef __attribute__((ext_vector_type(16))) __bf16 v16bf;
typedef __attribute__((ext_vector_type(8)))  float  v8f;

union FragB { v16bf v; unsigned u[8]; };

__device__ __forceinline__ unsigned short f2bf(float f) {
  unsigned u = __builtin_bit_cast(unsigned, f);
  unsigned r = u + 0x7FFFu + ((u >> 16) & 1u);   // round-to-nearest-even
  return (unsigned short)(r >> 16);
}
__device__ __forceinline__ unsigned pack2(float lo, float hi) {
  return (unsigned)f2bf(lo) | ((unsigned)f2bf(hi) << 16);
}

// ---------------- router: logits, softmax, top-2, slot-major dispatch lists ----
__global__ void __launch_bounds__(256)
router_kernel(const float* __restrict__ X, const float* __restrict__ Wr,
              float* __restrict__ logits, float* __restrict__ probs,
              int* __restrict__ expid, float* __restrict__ prv) {
  int t = blockIdx.x * 256 + threadIdx.x;
  float acc[NEXP];
  #pragma unroll
  for (int e = 0; e < NEXP; ++e) acc[e] = 0.f;
  const float4* xv = (const float4*)(X + (size_t)t * DMODEL);
  for (int d4 = 0; d4 < DMODEL / 4; ++d4) {
    float4 x = xv[d4];
    int d = d4 * 4;
    #pragma unroll
    for (int e = 0; e < NEXP; ++e)
      acc[e] += x.x * Wr[(size_t)d * NEXP + e] + x.y * Wr[(size_t)(d + 1) * NEXP + e] +
                x.z * Wr[(size_t)(d + 2) * NEXP + e] + x.w * Wr[(size_t)(d + 3) * NEXP + e];
  }
  float m = acc[0];
  #pragma unroll
  for (int e = 1; e < NEXP; ++e) m = fmaxf(m, acc[e]);
  float p[NEXP], s = 0.f;
  #pragma unroll
  for (int e = 0; e < NEXP; ++e) { p[e] = __expf(acc[e] - m); s += p[e]; }
  float inv = 1.f / s;
  #pragma unroll
  for (int e = 0; e < NEXP; ++e) {
    p[e] *= inv;
    logits[(size_t)t * NEXP + e] = acc[e];
    probs [(size_t)t * NEXP + e] = p[e];
  }
  int i0 = 0; float p0 = p[0];
  #pragma unroll
  for (int e = 1; e < NEXP; ++e) if (p[e] > p0) { p0 = p[e]; i0 = e; }
  int i1 = -1; float p1 = -1.f;
  #pragma unroll
  for (int e = 0; e < NEXP; ++e) if (e != i0 && p[e] > p1) { p1 = p[e]; i1 = e; }
  expid[t] = i0; expid[T_TOK + t] = i1;
  prv  [t] = p0; prv  [T_TOK + t] = p1;
}

// ---------------- capacity-constrained dispatch (deterministic block scan) ----
__global__ void __launch_bounds__(256)
dispatch_kernel(const int* __restrict__ expid, const float* __restrict__ prv,
                int* __restrict__ eidx, float* __restrict__ eprob) {
  __shared__ int s[256];
  const int e = blockIdx.x, tid = threadIdx.x;
  for (int i = tid; i < CAP; i += 256) { eidx[(size_t)e * CAP + i] = -1; eprob[(size_t)e * CAP + i] = 0.f; }
  __syncthreads();
  int running = 0;
  for (int bse = 0; bse < KTOP * T_TOK; bse += 256) {
    int j = bse + tid;
    int flag = (expid[j] == e) ? 1 : 0;
    s[tid] = flag;
    __syncthreads();
    for (int off = 1; off < 256; off <<= 1) {
      int v = s[tid];
      int add = (tid >= off) ? s[tid - off] : 0;
      __syncthreads();
      s[tid] = v + add;
      __syncthreads();
    }
    int incl = s[tid], total = s[255];
    int pos = running + incl - flag;  // exclusive prefix within slot-major order
    if (flag && pos < CAP) {
      eidx [(size_t)e * CAP + pos] = j & (T_TOK - 1);
      eprob[(size_t)e * CAP + pos] = prv[j];
    }
    running += total;
    __syncthreads();
  }
}

// ---------------- GEMM1: H = silu(X@Wg+bg) * (X@Wu+bu), bf16 WMMA -------------
__global__ void __launch_bounds__(128)
moe_gate_up(const float* __restrict__ X,
            const float* __restrict__ Wg, const float* __restrict__ bg,
            const float* __restrict__ Wu, const float* __restrict__ bu,
            unsigned short* __restrict__ H, int F, int rows,
            const int* __restrict__ idx, int base) {
  __shared__ unsigned aL[64 * 16];
  __shared__ unsigned gL[64 * 16];
  __shared__ unsigned uL[64 * 16];
  const int tid = threadIdx.x;
  const int lane = tid & 31, wave = tid >> 5;
  const int wm = (wave >> 1) * 32, wn = (wave & 1) * 32;
  const int kh = lane >> 4, nl = lane & 15;
  const int rowBase = blockIdx.x * 64, colBase = blockIdx.y * 64;

  v8f accg[2][2] = {}, accu[2][2] = {};

  for (int k0 = 0; k0 < DMODEL; k0 += 32) {
    // A tile: 64 gathered rows x 32 K, f32 -> packed bf16 pairs
    #pragma unroll
    for (int it = 0; it < 4; ++it) {
      int q = tid + it * 128;      // 0..511 float4 units
      int row = q >> 3;
      int kq = (q & 7) * 4;
      int rg = rowBase + row;
      float4 xv = make_float4(0.f, 0.f, 0.f, 0.f);
      if (rg < rows) {
        int tok = idx ? idx[rg] : (base + rg);
        if (tok < 0) tok = 0;
        xv = *(const float4*)(X + (size_t)tok * DMODEL + k0 + kq);
      }
      aL[row * 16 + (kq >> 1)]     = pack2(xv.x, xv.y);
      aL[row * 16 + (kq >> 1) + 1] = pack2(xv.z, xv.w);
    }
    // B tiles: 32x64 of Wg and Wu, packed along K pairs, laid out [n][kpair]
    #pragma unroll
    for (int it = 0; it < 8; ++it) {
      int q = tid + it * 128;      // 0..1023
      int kp = q >> 6, n = q & 63;
      size_t o0 = (size_t)(k0 + 2 * kp) * F + colBase + n;
      gL[n * 16 + kp] = pack2(Wg[o0], Wg[o0 + F]);
      uL[n * 16 + kp] = pack2(Wu[o0], Wu[o0 + F]);
    }
    if (k0 + 32 < DMODEL) {
      __builtin_prefetch(Wg + (size_t)(k0 + 32) * F + colBase, 0, 1);
      __builtin_prefetch(Wu + (size_t)(k0 + 32) * F + colBase, 0, 1);
    }
    __syncthreads();

    FragB af[2], gf[2], uf[2];
    #pragma unroll
    for (int mi = 0; mi < 2; ++mi)
      #pragma unroll
      for (int i = 0; i < 8; ++i) {
        int kp = (i >> 2) * 8 + kh * 4 + (i & 3);   // A layout: K pairs per lane-half
        af[mi].u[i] = aL[(wm + mi * 16 + nl) * 16 + kp];
      }
    #pragma unroll
    for (int ni = 0; ni < 2; ++ni)
      #pragma unroll
      for (int i = 0; i < 8; ++i) {
        int kp = (i >> 2) * 8 + kh * 4 + (i & 3);   // B layout: K pairs per lane-half
        gf[ni].u[i] = gL[(wn + ni * 16 + nl) * 16 + kp];
        uf[ni].u[i] = uL[(wn + ni * 16 + nl) * 16 + kp];
      }
    #pragma unroll
    for (int mi = 0; mi < 2; ++mi)
      #pragma unroll
      for (int ni = 0; ni < 2; ++ni) {
        accg[mi][ni] = __builtin_amdgcn_wmma_f32_16x16x32_bf16(
            false, af[mi].v, false, gf[ni].v, (short)0, accg[mi][ni], false, false);
        accu[mi][ni] = __builtin_amdgcn_wmma_f32_16x16x32_bf16(
            false, af[mi].v, false, uf[ni].v, (short)0, accu[mi][ni], false, false);
      }
    __syncthreads();
  }

  // epilogue: bias + SwiGLU, store H as bf16
  #pragma unroll
  for (int mi = 0; mi < 2; ++mi)
    #pragma unroll
    for (int ni = 0; ni < 2; ++ni) {
      int cg = colBase + wn + ni * 16 + nl;
      float bgv = bg[cg], buv = bu[cg];
      #pragma unroll
      for (int r = 0; r < 8; ++r) {
        int rg = rowBase + wm + mi * 16 + r + 8 * kh;
        if (rg < rows) {
          float g = accg[mi][ni][r] + bgv;
          float u = accu[mi][ni][r] + buv;
          float h = (g / (1.0f + __expf(-g))) * u;   // silu(g)*u
          H[(size_t)rg * F + cg] = f2bf(h);
        }
      }
    }
}

// ---------------- GEMM2: Y = H@Wo + bo, weighted combine into out -------------
__global__ void __launch_bounds__(128)
moe_down(const unsigned short* __restrict__ H,
         const float* __restrict__ Wo, const float* __restrict__ bo,
         float* __restrict__ out, int F, int rows,
         const int* __restrict__ idx, const float* __restrict__ prb, int base) {
  __shared__ unsigned aL[64 * 16];
  __shared__ unsigned bL[64 * 16];
  const int tid = threadIdx.x;
  const int lane = tid & 31, wave = tid >> 5;
  const int wm = (wave >> 1) * 32, wn = (wave & 1) * 32;
  const int kh = lane >> 4, nl = lane & 15;
  const int rowBase = blockIdx.x * 64, colBase = blockIdx.y * 64;

  v8f acc[2][2] = {};

  for (int k0 = 0; k0 < F; k0 += 32) {
    // A: H rows are already bf16; one u32 load yields a (k,k+1) pair
    #pragma unroll
    for (int it = 0; it < 8; ++it) {
      int q = tid + it * 128;
      int row = q >> 4, kp = q & 15;
      int rg = rowBase + row;
      unsigned v = 0;
      if (rg < rows) v = ((const unsigned*)H)[(size_t)rg * (F >> 1) + (k0 >> 1) + kp];
      aL[row * 16 + kp] = v;
    }
    // B: 32x64 of Wo (F x D), pack along K pairs
    #pragma unroll
    for (int it = 0; it < 8; ++it) {
      int q = tid + it * 128;
      int kp = q >> 6, n = q & 63;
      size_t o0 = (size_t)(k0 + 2 * kp) * DMODEL + colBase + n;
      bL[n * 16 + kp] = pack2(Wo[o0], Wo[o0 + DMODEL]);
    }
    if (k0 + 32 < F)
      __builtin_prefetch(Wo + (size_t)(k0 + 32) * DMODEL + colBase, 0, 1);
    __syncthreads();

    FragB af[2], bf[2];
    #pragma unroll
    for (int mi = 0; mi < 2; ++mi)
      #pragma unroll
      for (int i = 0; i < 8; ++i) {
        int kp = (i >> 2) * 8 + kh * 4 + (i & 3);
        af[mi].u[i] = aL[(wm + mi * 16 + nl) * 16 + kp];
      }
    #pragma unroll
    for (int ni = 0; ni < 2; ++ni)
      #pragma unroll
      for (int i = 0; i < 8; ++i) {
        int kp = (i >> 2) * 8 + kh * 4 + (i & 3);
        bf[ni].u[i] = bL[(wn + ni * 16 + nl) * 16 + kp];
      }
    #pragma unroll
    for (int mi = 0; mi < 2; ++mi)
      #pragma unroll
      for (int ni = 0; ni < 2; ++ni)
        acc[mi][ni] = __builtin_amdgcn_wmma_f32_16x16x32_bf16(
            false, af[mi].v, false, bf[ni].v, (short)0, acc[mi][ni], false, false);
    __syncthreads();
  }

  // epilogue: bias, routing weight, combine (each (token,col) touched once here;
  // kernels serialize on the stream, so no atomics needed)
  #pragma unroll
  for (int mi = 0; mi < 2; ++mi)
    #pragma unroll
    for (int ni = 0; ni < 2; ++ni) {
      int cg = colBase + wn + ni * 16 + nl;
      float bov = bo[cg];
      #pragma unroll
      for (int r = 0; r < 8; ++r) {
        int rg = rowBase + wm + mi * 16 + r + 8 * kh;
        if (rg < rows) {
          float y = acc[mi][ni][r] + bov;
          if (idx) {
            int tok = idx[rg];
            float w = prb[rg];
            if (tok >= 0 && w != 0.f) {
              float* p = out + (size_t)tok * DMODEL + cg;
              *p += y * w;
            }
          } else {
            float* p = out + (size_t)(base + rg) * DMODEL + cg;
            *p += y;
          }
        }
      }
    }
}

// ---------------------------------------------------------------------------
extern "C" void kernel_launch(void* const* d_in, const int* in_sizes, int n_in,
                              void* d_out, int out_size, void* d_ws, size_t ws_size,
                              hipStream_t stream) {
  (void)in_sizes; (void)n_in; (void)out_size;
  const float* X  = (const float*)d_in[0];
  const float* Wr = (const float*)d_in[1];
  const float *eWg[NEXP], *ebg[NEXP], *eWu[NEXP], *ebu[NEXP], *eWo[NEXP], *ebo[NEXP];
  for (int i = 0; i < NEXP; ++i) {
    eWg[i] = (const float*)d_in[2 + i];
    ebg[i] = (const float*)d_in[9 + i];
    eWu[i] = (const float*)d_in[16 + i];
    ebu[i] = (const float*)d_in[23 + i];
    eWo[i] = (const float*)d_in[30 + i];
    ebo[i] = (const float*)d_in[37 + i];
  }
  const float* sWg = (const float*)d_in[44];
  const float* sbg = (const float*)d_in[45];
  const float* sWu = (const float*)d_in[46];
  const float* sbu = (const float*)d_in[47];
  const float* sWo = (const float*)d_in[48];
  const float* sbo = (const float*)d_in[49];

  float* out    = (float*)d_out;
  float* logits = out + (size_t)T_TOK * DMODEL;
  float* probs  = logits + (size_t)T_TOK * NEXP;
  float* eprob  = probs + (size_t)T_TOK * NEXP;
  int*   eidx   = (int*)(eprob + (size_t)NEXP * CAP);

  int*            expid = (int*)d_ws;
  float*          prv   = (float*)((char*)d_ws + (size_t)KTOP * T_TOK * 4);
  unsigned short* Hws   = (unsigned short*)((char*)d_ws + 2 * (size_t)KTOP * T_TOK * 4);

  // chunk the H spill buffer to whatever scratch we were given
  size_t hbytes = (ws_size > (size_t)131072) ? ws_size - 131072 : 0;
  long long rmax = (long long)(hbytes / ((size_t)4096 * 2));
  int rpc = (int)((rmax / 64) * 64);
  if (rpc > 2048) rpc = 2048;
  if (rpc < 64)  rpc = 64;

  hipMemcpyAsync(out, X, (size_t)T_TOK * DMODEL * sizeof(float),
                 hipMemcpyDeviceToDevice, stream);
  router_kernel<<<T_TOK / 256, 256, 0, stream>>>(X, Wr, logits, probs, expid, prv);
  dispatch_kernel<<<NEXP, 256, 0, stream>>>(expid, prv, eidx, eprob);

  const int Fs[NEXP] = {4096, 3584, 4096, 3072, 4096, 3584, 3072};
  dim3 blk(128);
  for (int e = 0; e < NEXP; ++e) {
    int F = Fs[e];
    for (int b = 0; b < CAP; b += rpc) {
      int rows = (CAP - b < rpc) ? (CAP - b) : rpc;
      dim3 g1((rows + 63) / 64, F / 64);
      moe_gate_up<<<g1, blk, 0, stream>>>(X, eWg[e], ebg[e], eWu[e], ebu[e],
                                          Hws, F, rows, eidx + (size_t)e * CAP + b, 0);
      dim3 g2((rows + 63) / 64, DMODEL / 64);
      moe_down<<<g2, blk, 0, stream>>>(Hws, eWo[e], ebo[e], out, F, rows,
                                       eidx + (size_t)e * CAP + b,
                                       eprob + (size_t)e * CAP + b, 0);
    }
  }
  for (int b = 0; b < T_TOK; b += rpc) {
    int rows = (T_TOK - b < rpc) ? (T_TOK - b) : rpc;
    dim3 g1((rows + 63) / 64, 4096 / 64);
    moe_gate_up<<<g1, blk, 0, stream>>>(X, sWg, sbg, sWu, sbu, Hws, 4096, rows, nullptr, b);
    dim3 g2((rows + 63) / 64, DMODEL / 64);
    moe_down<<<g2, blk, 0, stream>>>(Hws, sWo, sbo, out, 4096, rows, nullptr, nullptr, b);
  }
}
